// ResidualStack_44727789420675
// MI455X (gfx1250) — matrix operand order repaired and checked by reading, not compiled, not run on hardware
//
#include <hip/hip_runtime.h>
#include <math.h>

// ---------------------------------------------------------------------------
// WaveNet residual stack for MI455X (gfx1250), wave32 + WMMA bf16.
//   B=8, CHAN_RES=64, CHAN_SKIP=128, T=8192, 40 layers, SKIP_SIZE=4096.
// One kernel launch per layer; activations ping-pong in d_ws (fp32, stride T).
// ---------------------------------------------------------------------------

typedef __attribute__((ext_vector_type(16))) __bf16   v16bf;
typedef __attribute__((ext_vector_type(8)))  float    v8f;
typedef __attribute__((ext_vector_type(4)))  unsigned uint4v;

#define WMMA_BF16(A, B, C) \
  __builtin_amdgcn_wmma_f32_16x16x32_bf16(false, (A), false, (B), (short)0, (C), false, false)

#define TSTRIDE 8192   // fixed time stride of activation buffers
#define TB      128    // timesteps per block

__device__ __forceinline__ __bf16 f2bf(float f) {
  union { float f; unsigned u; } a; a.f = f;
  unsigned u = a.u + 0x7fffu + ((a.u >> 16) & 1u);   // round-to-nearest-even
  union { unsigned short s; __bf16 h; } b;
  b.s = (unsigned short)(u >> 16);
  return b.h;
}

__device__ __forceinline__ unsigned pack2(float f0, float f1) {
  union { __bf16 h[2]; unsigned u; } p;
  p.h[0] = f2bf(f0); p.h[1] = f2bf(f1);
  return p.u;
}

__device__ __forceinline__ float fast_rcp(float x) {
#if __has_builtin(__builtin_amdgcn_rcpf)
  return __builtin_amdgcn_rcpf(x);
#else
  return 1.0f / x;
#endif
}

// gate(y) = tanh(y) * sigmoid(y), using gfx1250 TRANS ops (v_tanh/v_exp/v_rcp)
// instead of libm polynomials.  Stable for all y: only exp(-|y|) is formed.
__device__ __forceinline__ float gatef(float y) {
  float ay = fabsf(y);
  float e  = __expf(-ay);                 // v_exp_f32, e in (0,1]
#if __has_builtin(__builtin_amdgcn_tanhf)
  float th = __builtin_amdgcn_tanhf(ay);  // v_tanh_f32
#else
  float e2 = e * e;
  float th = (1.0f - e2) * fast_rcp(1.0f + e2);
#endif
  float sg = fast_rcp(1.0f + e);          // sigmoid(|y|)
  float t  = (y >= 0.0f) ? th : -th;      // tanh is odd
  float s  = (y >= 0.0f) ? sg : e * sg;   // sigmoid(-|y|) = e/(1+e)
  return t * s;
}

// A-fragment (16x32 bf16, M x K) from LDS weight matrix W[rows][64], row o.
// 16-bit A layout: lanes 0-15 hold K {k0..k0+7, k0+16..k0+23},
//                  lanes 16-31 hold K {k0+8..k0+15, k0+24..k0+31}.
__device__ __forceinline__ v16bf load_A(const __bf16* W, int o, int k0, int half) {
  const __bf16* row = W + o * 64 + k0 + half * 8;
  union { v16bf v; uint4v q[2]; } u;
  u.q[0] = *(const uint4v*)(row);
  u.q[1] = *(const uint4v*)(row + 16);
  return u.v;
}

// B-fragment (32x16 bf16, K x N) from time-major LDS tile X[t][64].
// Lane holds column N = lane&15 at time t; K = {k0+half*8..+7, k0+16+half*8..+7}.
__device__ __forceinline__ v16bf load_B(const __bf16* Xtm, int t, int k0, int half) {
  const __bf16* p = Xtm + t * 64 + k0 + half * 8;
  union { v16bf v; uint4v q[2]; } u;
  u.q[0] = *(const uint4v*)(p);
  u.q[1] = *(const uint4v*)(p + 16);
  return u.v;
}

// LDS layout (bf16): xa[128][64] xb[128][64] g[128][64] w0[64][64] w1[64][64]
//                    wres[64][64] wskip[128][64]  -> 45056 elems = 90112 B
#define SMEM_BYTES 90112

__global__ __launch_bounds__(256)
void wavenet_layer_kernel(const float* __restrict__ xin,  float* __restrict__ xout,
                          const float* __restrict__ w_dil, const float* __restrict__ w_res,
                          const float* __restrict__ b_res, const float* __restrict__ w_skip,
                          const float* __restrict__ b_skip, float* __restrict__ skip_out,
                          int layer, int dil, int T_in, int T_out, int blocks_t)
{
  extern __shared__ __align__(32) char smem[];
  __bf16* lds_xa = (__bf16*)smem;              // [128][64] time-major
  __bf16* lds_xb = lds_xa + 128 * 64;          // [128][64]
  __bf16* lds_g  = lds_xb + 128 * 64;          // [128][64]
  __bf16* lds_w0 = lds_g  + 128 * 64;          // [64][64]
  __bf16* lds_w1 = lds_w0 + 64 * 64;           // [64][64]
  __bf16* lds_wr = lds_w1 + 64 * 64;           // [64][64]
  __bf16* lds_ws = lds_wr + 64 * 64;           // [128][64]

  const int tid = threadIdx.x;
  const int b   = blockIdx.x / blocks_t;
  const int t0  = (blockIdx.x % blocks_t) * TB;

  // ---- stage weights for this layer as bf16 ----
  {
    const float* wd = w_dil + (size_t)layer * 64 * 64 * 2;   // [o][i][k]
    const float* wr = w_res + (size_t)layer * 64 * 64;       // [o][i]
    for (int e = tid; e < 64 * 64; e += 256) {
      lds_w0[e] = f2bf(wd[e * 2 + 0]);
      lds_w1[e] = f2bf(wd[e * 2 + 1]);
      lds_wr[e] = f2bf(wr[e]);
    }
    const float* wk = w_skip + (size_t)layer * 128 * 64;     // [o][i]
    for (int e = tid; e < 128 * 64; e += 256)
      lds_ws[e] = f2bf(wk[e]);
  }

  // ---- stage input tiles (both conv taps) time-major ----
  {
    const int ch   = tid >> 2;   // 0..63
    const int part = tid & 3;    // 0..3, 32 timesteps each
    const float* rowp = xin + ((size_t)b * 64 + ch) * TSTRIDE;
    for (int k = 0; k < 32; ++k) {
      int tl = part * 32 + k;
      int ta = t0 + tl;
      int tb = ta + dil;
      float va = (ta < T_in) ? rowp[ta] : 0.0f;
      float vb = (tb < T_in) ? rowp[tb] : 0.0f;
      lds_xa[tl * 64 + ch] = f2bf(va);
      lds_xb[tl * 64 + ch] = f2bf(vb);
    }
  }
  __syncthreads();

  const int wv   = tid >> 5;
  const int lane = tid & 31;
  const int half = lane >> 4;
  const int n    = lane & 15;
  const int tl   = wv * 16;        // this wave's 16-timestep strip
  const int tcol = tl + n;         // lane's B/D column (local time)

  // ---- dilated causal conv: Y = W0*Xa + W1*Xb  (M=64, N=16, K=64) ----
  v8f acc[4];
  {
    v16bf ba0 = load_B(lds_xa, tcol, 0,  half);
    v16bf ba1 = load_B(lds_xa, tcol, 32, half);
    v16bf bb0 = load_B(lds_xb, tcol, 0,  half);
    v16bf bb1 = load_B(lds_xb, tcol, 32, half);
    #pragma unroll
    for (int mt = 0; mt < 4; ++mt) {
      const int o = mt * 16 + n;   // A row index = lane&15
      v8f c = {0.f,0.f,0.f,0.f,0.f,0.f,0.f,0.f};
      c = WMMA_BF16(load_A(lds_w0, o, 0,  half), ba0, c);
      c = WMMA_BF16(load_A(lds_w0, o, 32, half), ba1, c);
      c = WMMA_BF16(load_A(lds_w1, o, 0,  half), bb0, c);
      c = WMMA_BF16(load_A(lds_w1, o, 32, half), bb1, c);
      acc[mt] = c;
    }
  }

  // ---- gate = tanh(y)*sigmoid(y); write G to LDS (time-major bf16) ----
  // D layout: VGPR r -> channel row (r + 8*half) within tile, column n.
  #pragma unroll
  for (int mt = 0; mt < 4; ++mt) {
    #pragma unroll
    for (int r = 0; r < 8; r += 2) {
      float g0 = gatef(acc[mt][r]);
      float g1 = gatef(acc[mt][r + 1]);
      int ch = mt * 16 + half * 8 + r;
      *(unsigned*)(lds_g + tcol * 64 + ch) = pack2(g0, g1);
    }
  }
  __syncthreads();

  v16bf bg0 = load_B(lds_g, tcol, 0,  half);
  v16bf bg1 = load_B(lds_g, tcol, 32, half);

  const int  tglob = t0 + tcol;
  const bool valid = (tglob < T_out);

  // ---- residual: out = Wres*G + b_res + x[t+d]  (fp32 state in global) ----
  #pragma unroll
  for (int mt = 0; mt < 4; ++mt) {
    v8f c = {0.f,0.f,0.f,0.f,0.f,0.f,0.f,0.f};
    c = WMMA_BF16(load_A(lds_wr, mt * 16 + n, 0,  half), bg0, c);
    c = WMMA_BF16(load_A(lds_wr, mt * 16 + n, 32, half), bg1, c);
    if (valid) {
      #pragma unroll
      for (int r = 0; r < 8; ++r) {
        int o = mt * 16 + half * 8 + r;
        size_t base = ((size_t)b * 64 + o) * TSTRIDE;
        xout[base + tglob] = c[r] + b_res[layer * 64 + o] + xin[base + tglob + dil];
      }
    }
  }

  // ---- skip: Wskip*G + b_skip, keep last 4096 timesteps ----
  const int sstart = T_out - 4096;
  #pragma unroll
  for (int mt = 0; mt < 8; ++mt) {
    v8f c = {0.f,0.f,0.f,0.f,0.f,0.f,0.f,0.f};
    c = WMMA_BF16(load_A(lds_ws, mt * 16 + n, 0,  half), bg0, c);
    c = WMMA_BF16(load_A(lds_ws, mt * 16 + n, 32, half), bg1, c);
    const int s = tglob - sstart;
    if (valid && s >= 0) {
      #pragma unroll
      for (int r = 0; r < 8; ++r) {
        int o = mt * 16 + half * 8 + r;
        size_t idx = ((((size_t)layer * 8 + b) * 128) + o) * 4096 + (size_t)s;
        skip_out[idx] = c[r] + b_skip[layer * 128 + o];
      }
    }
  }
}

// ---------------------------------------------------------------------------
extern "C" void kernel_launch(void* const* d_in, const int* in_sizes, int n_in,
                              void* d_out, int out_size, void* d_ws, size_t ws_size,
                              hipStream_t stream)
{
  (void)in_sizes; (void)n_in; (void)out_size; (void)ws_size;

  static const int DIL[40] = {
    1,2,4,8,16,32,64,128,256,512,
    1,2,4,8,16,32,64,128,256,512,
    1,2,4,8,16,32,64,128,256,512,
    1,2,4,8,16,32,64,128,256,512 };

  const float* x      = (const float*)d_in[0];
  const float* w_dil  = (const float*)d_in[1];
  const float* w_res  = (const float*)d_in[2];
  const float* b_res  = (const float*)d_in[3];
  const float* w_skip = (const float*)d_in[4];
  const float* b_skip = (const float*)d_in[5];
  float* skip_out = (float*)d_out;

  // ping-pong fp32 activation buffers, fixed stride TSTRIDE
  float* bufA = (float*)d_ws;
  float* bufB = bufA + (size_t)8 * 64 * TSTRIDE;

  int T = TSTRIDE;
  for (int i = 0; i < 40; ++i) {
    const int dd    = DIL[i];
    const int T_out = T - dd;
    const float* inp = (i == 0) ? x : ((i & 1) ? bufA : bufB);
    float*       outp = (i & 1) ? bufB : bufA;
    const int bt = (T_out + TB - 1) / TB;
    dim3 grid(bt * 8), block(256);
    wavenet_layer_kernel<<<grid, block, SMEM_BYTES, stream>>>(
        inp, outp, w_dil, w_res, b_res, w_skip, b_skip, skip_out,
        i, dd, T, T_out, bt);
    T = T_out;
  }
}
